// MLSoftmaxDecoder_90563680403701
// MI455X (gfx1250) — matrix operand order repaired and checked by reading, not compiled, not run on hardware
//
#include <hip/hip_runtime.h>
#include <hip/hip_bf16.h>
#include <cstdint>
#include <cstddef>

typedef _Float16 half_t;
typedef __attribute__((ext_vector_type(16))) _Float16 v16h;
typedef __attribute__((ext_vector_type(8)))  float    v8f;

#define B_  16
#define T_  500
#define E_  1024
#define D_  1024
#define LC_ 80
#define LB_ 40
#define VC_ 4000
#define VB_ 2000
#define EC_ 128
#define EB_ 1000

__device__ __forceinline__ float sigm(float x) { return 1.0f / (1.0f + __expf(-x)); }

// ===================== generic WMMA GEMM =====================
// C[M,N] = A[M,K](f16,row-major,lda) x Bpacked[K,N](f16, WMMA-fragment tile layout) + bias
// Bpacked layout: tiles (K/32 x N/16), each tile = 32 lanes x 16 halves contiguous.
// swapRB: output row m=(l*rowsInner+b) remapped to b*rowsOuter+l (writes [B,L,*] directly).
__global__ void k_wmma_gemm(const half_t* __restrict__ A, int lda,
                            const half_t* __restrict__ Bp,
                            const float* __restrict__ bias,
                            float* __restrict__ C, long ldc,
                            int M, int N, int K,
                            int swapRB, int rowsInner, int rowsOuter)
{
  const int wavesPerBlock = blockDim.x >> 5;
  int wave = blockIdx.x * wavesPerBlock + (threadIdx.x >> 5);
  int lane = threadIdx.x & 31;
  int ntiles = N >> 4;
  int tiles  = (M >> 4) * ntiles;
  if (wave >= tiles) return;
  int mt = wave / ntiles;
  int nt = wave - mt * ntiles;
  int lm = lane & 15;
  int hi = lane >> 4;

  const half_t* arow = A + (size_t)(mt * 16 + lm) * lda + hi * 8;
  const v16h*  bbase = (const v16h*)Bp + (size_t)nt * 32 + lane;
  const int    bstep = ntiles * 32;          // v16h units per K-tile

  v8f acc = {};
  int ktn = K >> 5;
  for (int kt = 0; kt < ktn; ++kt) {
    const half_t* ap = arow + kt * 32;
    v16h a;
#pragma unroll
    for (int v = 0; v < 8; ++v) {
      int k0 = (v < 4) ? (2 * v) : (16 + 2 * (v - 4));
      a[2 * v]     = ap[k0];
      a[2 * v + 1] = ap[k0 + 1];
    }
    v16h b = bbase[(size_t)kt * bstep];
    __builtin_prefetch((const char*)&bbase[(size_t)(kt + 1) * bstep], 0, 1); // global_prefetch_b8
    acc = __builtin_amdgcn_wmma_f32_16x16x32_f16(false, a, false, b, (short)0, acc, false, false);
  }

  int n  = nt * 16 + lm;
  float bv = bias ? bias[n] : 0.0f;
  int mbase = mt * 16 + hi * 8;
#pragma unroll
  for (int r = 0; r < 8; ++r) {
    int mm = mbase + r;
    float v = acc[r] + bv;
    size_t orow = swapRB ? ((size_t)(mm % rowsInner) * rowsOuter + (mm / rowsInner))
                         : (size_t)mm;
    C[orow * ldc + n] = v;
  }
}

// ===================== M=16 WMMA GEMM with async-LDS-staged A =====================
// For the recurrent-step GEMMs: A is a single 16xK activation shared by every
// N-tile. Stage it into LDS once per workgroup with the CDNA5 async global->LDS
// path (ASYNCcnt / s_wait_asynccnt), then build WMMA A-fragments from LDS.
// Requires lda == K (contiguous A), K multiple of 128.
__global__ void k_wmma_gemm_m16(const half_t* __restrict__ A,
                                const half_t* __restrict__ Bp,
                                const float* __restrict__ bias,
                                float* __restrict__ C, long ldc,
                                int N, int K)
{
  extern __shared__ half_t As[];                 // [16][K]
  // --- async stage: 16-byte chunk per lane per issue; 16*K halves total ---
  int chunks = (16 * K) >> 3;                    // # of 8-half (16B) chunks
  for (int c = threadIdx.x; c < chunks; c += blockDim.x) {
    unsigned           ldsa = (unsigned)(uintptr_t)(As + (size_t)c * 8);
    unsigned long long ga   = (unsigned long long)(uintptr_t)(A + (size_t)c * 8);
    asm volatile("global_load_async_to_lds_b128 %0, %1, off"
                 :: "v"(ldsa), "v"(ga) : "memory");
  }
  asm volatile("s_wait_asynccnt 0x0" ::: "memory");
  __syncthreads();

  int lane = threadIdx.x & 31;
  int nt = blockIdx.x * (blockDim.x >> 5) + (threadIdx.x >> 5);
  int ntiles = N >> 4;
  if (nt >= ntiles) return;
  int lm = lane & 15, hi = lane >> 4;

  const half_t* arow = As + (size_t)lm * K + hi * 8;
  const v16h*  bbase = (const v16h*)Bp + (size_t)nt * 32 + lane;
  const int    bstep = ntiles * 32;

  v8f acc = {};
  int ktn = K >> 5;
  for (int kt = 0; kt < ktn; ++kt) {
    const half_t* ap = arow + kt * 32;
    v16h a;
#pragma unroll
    for (int v = 0; v < 8; ++v) {
      int k0 = (v < 4) ? (2 * v) : (16 + 2 * (v - 4));
      a[2 * v]     = ap[k0];               // ds_load_b128 x2 per K-tile
      a[2 * v + 1] = ap[k0 + 1];
    }
    v16h b = bbase[(size_t)kt * bstep];
    __builtin_prefetch((const char*)&bbase[(size_t)(kt + 1) * bstep], 0, 1);
    acc = __builtin_amdgcn_wmma_f32_16x16x32_f16(false, a, false, b, (short)0, acc, false, false);
  }

  int n = nt * 16 + lm;
  float bv = bias ? bias[n] : 0.0f;
  int mbase = hi * 8;
#pragma unroll
  for (int r = 0; r < 8; ++r)
    C[(size_t)(mbase + r) * ldc + n] = acc[r] + bv;
}

// ===================== weight staging / packing =====================
__global__ void k_padcopy(const float* __restrict__ src, int rows, int cols,
                          half_t* __restrict__ dst, int ld, int ro, int co)
{
  long i = blockIdx.x * (long)blockDim.x + threadIdx.x;
  if (i >= (long)rows * cols) return;
  long r = i / cols; int c = (int)(i - r * cols);
  dst[(ro + r) * (long)ld + co + c] = (half_t)src[i];
}

__global__ void k_pack_b(const half_t* __restrict__ W, half_t* __restrict__ out, int Kp, int Np)
{
  long i = blockIdx.x * (long)blockDim.x + threadIdx.x;
  if (i >= (long)Kp * Np) return;
  int ntiles = Np >> 4;
  long tile = i >> 9;            // 512 halves per tile
  int  r    = (int)(i & 511);
  int  lane = r >> 4, e = r & 15;
  int  kt = (int)(tile / ntiles), nt = (int)(tile % ntiles);
  int  k = kt * 32 + (lane >> 4) * 8 + (e < 8 ? e : 8 + e);
  int  n = nt * 16 + (lane & 15);
  out[i] = W[(long)k * Np + n];
}

// ===================== elementwise helpers =====================
__global__ void k_copy_f32(float* __restrict__ d, const float* __restrict__ s, int n)
{ int i = blockIdx.x * blockDim.x + threadIdx.x; if (i < n) d[i] = s[i]; }

__global__ void k_cast16(const float* __restrict__ s, half_t* __restrict__ d, long n)
{ long i = blockIdx.x * (long)blockDim.x + threadIdx.x; if (i < n) d[i] = (half_t)s[i]; }

__global__ void k_relu16(const float* __restrict__ s, half_t* __restrict__ d, long n)
{ long i = blockIdx.x * (long)blockDim.x + threadIdx.x; if (i < n) d[i] = (half_t)fmaxf(s[i], 0.0f); }

__global__ void k_add2(float* __restrict__ d, const float* __restrict__ a,
                       const float* __restrict__ b, long n)
{ long i = blockIdx.x * (long)blockDim.x + threadIdx.x; if (i < n) d[i] = a[i] + b[i]; }

__global__ void k_maxout2(const float* __restrict__ s, half_t* __restrict__ d, int rows, int hc)
{
  long i = blockIdx.x * (long)blockDim.x + threadIdx.x;
  if (i >= (long)rows * hc) return;
  long r = i / hc; int j = (int)(i - r * hc);
  const float* p = s + r * 2L * hc + 2 * j;
  d[i] = (half_t)fmaxf(p[0], p[1]);
}

// ===================== embeddings / init =====================
__global__ void k_embed_c(const float* __restrict__ tab, const int* __restrict__ ids,
                          half_t* __restrict__ out)
{
  int i = blockIdx.x * blockDim.x + threadIdx.x;
  if (i >= LC_ * B_ * EC_) return;
  int e = i & (EC_ - 1); int rb = i >> 7; int b = rb & 15; int l = rb >> 4;
  out[i] = (half_t)tab[(long)ids[b * LC_ + l] * EC_ + e];
}

__global__ void k_embed_b(const float* __restrict__ tab, const int* __restrict__ ids,
                          half_t* __restrict__ out)  // padded 1000 -> 1024
{
  int i = blockIdx.x * blockDim.x + threadIdx.x;
  if (i >= LB_ * B_ * 1024) return;
  int e = i & 1023; int rb = i >> 10; int b = rb & 15; int l = rb >> 4;
  out[i] = (e < EB_) ? (half_t)tab[(long)ids[b * LB_ + l] * EB_ + e] : (half_t)0.0f;
}

__global__ void k_encmean(const float* __restrict__ hs, const int* __restrict__ hlens,
                          float* __restrict__ m32, half_t* __restrict__ m16)
{
  int i = blockIdx.x * blockDim.x + threadIdx.x;
  if (i >= B_ * E_) return;
  int b = i >> 10, e = i & 1023;
  int len = hlens[b];
  const float* p = hs + (long)b * T_ * E_ + e;
  float s = 0.f;
  for (int t = 0; t < len; ++t) s += p[(long)t * E_];
  s /= (float)len;
  m32[i] = s; m16[i] = (half_t)s;
}

// ===================== GRU step pieces =====================
__global__ void k_gru1(const float* __restrict__ xr, const float* __restrict__ Y1,
                       const float* __restrict__ st, float* __restrict__ h1,
                       half_t* __restrict__ hc16)
{
  int i = blockIdx.x * blockDim.x + threadIdx.x;
  if (i >= B_ * D_) return;
  int b = i >> 10, j = i & 1023;
  const float* x = xr + b * 3072;
  const float* y = Y1 + b * 3072;
  float r1 = sigm(x[j] + y[j]);
  float u1 = sigm(x[1024 + j] + y[1024 + j]);
  float h  = tanhf(x[2048 + j] + y[2048 + j] * r1);
  h = u1 * st[i] + (1.f - u1) * h;
  h1[i] = h;
  hc16[b * 2048 + j] = (half_t)h;
}

__global__ void k_gru2(const float* __restrict__ Y2, const float* __restrict__ h1,
                       float* __restrict__ st, float* __restrict__ hsOut)
{
  int i = blockIdx.x * blockDim.x + threadIdx.x;
  if (i >= B_ * D_) return;
  int b = i >> 10, j = i & 1023;
  const float* y = Y2 + b * 4096;
  float r2 = sigm(y[j]);
  float u2 = sigm(y[1024 + j]);
  float h  = tanhf(y[2048 + j] * r2 + y[3072 + j]);
  float h2 = u2 * h1[i] + (1.f - u2) * h;
  st[i] = h2; hsOut[i] = h2;
}

// ===================== attention (wave32 VALU) =====================
__global__ void k_att_scores(const float* __restrict__ h1, const float* __restrict__ mem,
                             long sB, long sT, int T, float scale, float* __restrict__ scores)
{
  int wave = (int)((blockIdx.x * (long)blockDim.x + threadIdx.x) >> 5);
  int lane = threadIdx.x & 31;
  int b = wave / T, t = wave % T;
  if (b >= B_) return;
  const float* hp = h1 + b * 1024;
  const float* mp = mem + (size_t)b * sB + (size_t)t * sT;
  float p = 0.f;
  for (int d = lane; d < 1024; d += 32) p += hp[d] * mp[d];
  for (int off = 16; off; off >>= 1) p += __shfl_down(p, off, 32);
  if (lane == 0) scores[b * 512 + t] = p * scale;
}

__global__ void k_att_softmax(const float* __restrict__ scores, const int* __restrict__ lens,
                              int T, float* __restrict__ alpha)
{
  int b = blockIdx.x, lane = threadIdx.x;
  int len = lens ? lens[b] : T;
  float mx = -1e30f;
  for (int t = lane; t < len; t += 32) mx = fmaxf(mx, scores[b * 512 + t]);
  for (int off = 16; off; off >>= 1) mx = fmaxf(mx, __shfl_xor(mx, off, 32));
  float sum = 0.f;
  for (int t = lane; t < len; t += 32) sum += __expf(scores[b * 512 + t] - mx);
  for (int off = 16; off; off >>= 1) sum += __shfl_xor(sum, off, 32);
  float inv = 1.0f / sum;
  for (int t = lane; t < T; t += 32)
    alpha[b * 512 + t] = (t < len) ? __expf(scores[b * 512 + t] - mx) * inv : 0.0f;
}

__global__ void k_att_ctx(const float* __restrict__ alpha, const float* __restrict__ mem,
                          long sB, long sT, int T,
                          float* __restrict__ ctxOut, half_t* __restrict__ hc16)
{
  int i = blockIdx.x * blockDim.x + threadIdx.x;
  if (i >= B_ * E_) return;
  int b = i >> 10, d = i & 1023;
  const float* mp = mem + (size_t)b * sB + d;
  const float* ap = alpha + b * 512;
  float s = 0.f;
  for (int t = 0; t < T; ++t) s += ap[t] * mp[(size_t)t * sT];
  ctxOut[i] = s;
  hc16[b * 2048 + 1024 + d] = (half_t)s;
}

// ===================== classifier A-matrix builders =====================
__global__ void k_cat_c(const float* __restrict__ hs_c, const half_t* __restrict__ embc,
                        const float* __restrict__ ctx_c, half_t* __restrict__ out)
{
  long i = blockIdx.x * (long)blockDim.x + threadIdx.x;
  if (i >= (long)LC_ * B_ * 2176) return;
  long r = i / 2176; int c = (int)(i - r * 2176);
  half_t v;
  if (c < 1024)       v = (half_t)hs_c[r * 1024 + c];
  else if (c < 1152)  v = embc[r * 128 + (c - 1024)];
  else                v = (half_t)ctx_c[r * 1024 + (c - 1152)];
  out[i] = v;
}

__global__ void k_cat_b(const float* __restrict__ hs_b, const float* __restrict__ g,
                        const float* __restrict__ hs_g, const half_t* __restrict__ embb,
                        const float* __restrict__ ctx_b, const float* __restrict__ ctx_g,
                        half_t* __restrict__ out)
{
  long i = blockIdx.x * (long)blockDim.x + threadIdx.x;
  if (i >= (long)LB_ * B_ * 5120) return;
  long r = i / 5120; int c = (int)(i - r * 5120);
  half_t v;
  if (c < 1024)       v = (half_t)hs_b[r * 1024 + c];
  else if (c < 2048) { int j = c - 1024; v = (half_t)(sigm(g[r * 1024 + j]) * hs_g[r * 1024 + j]); }
  else if (c < 3072)  v = embb[r * 1024 + (c - 2048)];
  else if (c < 4096)  v = (half_t)ctx_b[r * 1024 + (c - 3072)];
  else                v = (half_t)ctx_g[r * 1024 + (c - 4096)];
  out[i] = v;
}

// ===================== host helpers =====================
static inline void run_gemm(hipStream_t s, const half_t* A, int lda, const half_t* Bp,
                            const float* bias, float* C, long ldc,
                            int M, int N, int K, int swapRB = 0, int inner = 16, int outer = 0)
{
  int tiles = (M / 16) * (N / 16);
  int blocks = (tiles + 7) / 8;
  k_wmma_gemm<<<blocks, 256, 0, s>>>(A, lda, Bp, bias, C, ldc, M, N, K, swapRB, inner, outer);
}
static inline void run_gemm_m16(hipStream_t s, const half_t* A, const half_t* Bp,
                                const float* bias, float* C, long ldc, int N, int K)
{
  int ntiles = N / 16;
  int blocks = (ntiles + 7) / 8;
  size_t shmem = (size_t)16 * K * sizeof(half_t);   // <= 64 KB (K <= 2048)
  k_wmma_gemm_m16<<<blocks, 256, shmem, s>>>(A, Bp, bias, C, ldc, N, K);
}
static inline void padc(hipStream_t s, const float* src, int rows, int cols,
                        half_t* stag, int ld, int ro, int co)
{
  long n = (long)rows * cols;
  k_padcopy<<<(int)((n + 255) / 256), 256, 0, s>>>(src, rows, cols, stag, ld, ro, co);
}
static inline void packW(hipStream_t s, const half_t* stag, half_t* dst, int Kp, int Np)
{
  long n = (long)Kp * Np;
  k_pack_b<<<(int)((n + 255) / 256), 256, 0, s>>>(stag, dst, Kp, Np);
}
static inline void copyf(hipStream_t s, float* dst, const float* src, int n)
{ k_copy_f32<<<(n + 255) / 256, 256, 0, s>>>(dst, src, n); }

static void run_stack(hipStream_t s, int L,
                      const float* mem, long sB, long sT, int T, const int* lens,
                      const float* xW, const half_t* W1p, const half_t* W2p, const float* bias2,
                      const float* s0, float* state, half_t* sA16, half_t* hc16,
                      float* Y1, float* Y2, float* h1buf, float* scores, float* alpha,
                      float* hsStore, float* ctxStore)
{
  copyf(s, state, s0, B_ * D_);
  for (int l = 0; l < L; ++l) {
    k_cast16<<<64, 256, 0, s>>>(state, sA16, (long)B_ * D_);
    run_gemm_m16(s, sA16, W1p, nullptr, Y1, 3072, 3072, 1024);
    k_gru1<<<64, 256, 0, s>>>(xW + (size_t)l * B_ * 3072, Y1, state, h1buf, hc16);
    int swaves = B_ * T;
    k_att_scores<<<(swaves + 7) / 8, 256, 0, s>>>(h1buf, mem, sB, sT, T, 0.03125f, scores);
    k_att_softmax<<<16, 32, 0, s>>>(scores, lens, T, alpha);
    k_att_ctx<<<64, 256, 0, s>>>(alpha, mem, sB, sT, T, ctxStore + (size_t)l * B_ * E_, hc16);
    run_gemm_m16(s, hc16, W2p, bias2, Y2, 4096, 4096, 2048);
    k_gru2<<<64, 256, 0, s>>>(Y2, h1buf, state, hsStore + (size_t)l * B_ * D_);
  }
}

extern "C" void kernel_launch(void* const* d_in, const int* in_sizes, int n_in,
                              void* d_out, int out_size, void* d_ws, size_t ws_size,
                              hipStream_t stream)
{
  (void)in_sizes; (void)n_in; (void)out_size; (void)ws_size;
  auto F = [&](int i) { return (const float*)d_in[i]; };
  auto I = [&](int i) { return (const int*)d_in[i]; };

  // input order: hs_pad, hlens, ys_char, ys_bpe, then params in insertion order
  const float* hs_pad = F(0);
  const int*   hlens  = I(1);
  const int*   ysc    = I(2);
  const int*   ysb    = I(3);
  // params: 4 c_emb, 5 b_emb, 6..17 rc, 18..29 rb, 30..41 rg,
  // 42 c_init 43 c_binit 44 b_init 45 b_binit,
  // 46 cc_sW 47 cc_bs 48 cc_yW 49 cc_eW 50 cc_out 51 cc_bout,
  // 52 b2c_W1 53 b2c_b1 54 b2c_W2 55 b2c_b2, 56 comb_W 57 comb_b,
  // 58 cb_sW 59 cb_bs 60 cb_yW 61 cb_eW 62 cb_out 63 cb_bout

  // ---------------- workspace carve-out ----------------
  char* ws = (char*)d_ws; size_t off = 0;
  auto alloc = [&](size_t bytes) { char* p = ws + off; off += (bytes + 255) & ~(size_t)255; return p; };
  auto allocH = [&](size_t n) { return (half_t*)alloc(n * sizeof(half_t)); };
  auto allocF = [&](size_t n) { return (float*)alloc(n * sizeof(float)); };

  half_t* staging = allocH((size_t)5120 * 2048);   // max padded weight
  half_t* cinit_p = allocH((size_t)1024 * 1024);
  half_t* binit_p = allocH((size_t)1024 * 1024);

  half_t* Wx_p[3]; half_t* W1_p[3]; half_t* W2_p[3]; float* xbias[3]; float* bias2[3];
  const int sbase[3] = {6, 18, 30};
  const int ein[3] = {EC_, EB_, EB_};
  const int einP[3] = {128, 1024, 1024};
  for (int k = 0; k < 3; ++k) {
    Wx_p[k]  = allocH((size_t)einP[k] * 3072);
    W1_p[k]  = allocH((size_t)1024 * 3072);
    W2_p[k]  = allocH((size_t)2048 * 4096);
    xbias[k] = allocF(3072);
    bias2[k] = allocF(4096);
  }
  half_t* cc_p    = allocH((size_t)2176 * 1024);
  half_t* ccout_p = allocH((size_t)512 * 4000);
  half_t* b2c1_p  = allocH((size_t)1024 * 1024);
  half_t* b2c2_p  = allocH((size_t)1024 * 1024);
  half_t* comb_p  = allocH((size_t)1024 * 1024);
  half_t* cb_p    = allocH((size_t)5120 * 2048);
  half_t* cbout_p = allocH((size_t)1024 * 2000);

  half_t* embc16 = allocH((size_t)LC_ * B_ * EC_);
  half_t* embb16 = allocH((size_t)LB_ * B_ * 1024);
  float*  mean32 = allocF(B_ * E_); half_t* mean16 = allocH(B_ * E_);
  float*  s0c = allocF(B_ * D_);    float* s0b = allocF(B_ * D_);
  float*  xWc = allocF((size_t)LC_ * B_ * 3072);
  float*  xWb = allocF((size_t)LB_ * B_ * 3072);
  float*  xWg = allocF((size_t)LB_ * B_ * 3072);

  float*  state = allocF(B_ * D_);
  half_t* sA16  = allocH(B_ * D_);
  half_t* hc16  = allocH(B_ * 2048);
  float*  Y1    = allocF(B_ * 3072);
  float*  Y2    = allocF(B_ * 4096);
  float*  h1buf = allocF(B_ * D_);
  float*  scores = allocF(B_ * 512);
  float*  alphab = allocF(B_ * 512);

  float* hs_c = allocF((size_t)LC_ * B_ * D_); float* ctx_c = allocF((size_t)LC_ * B_ * E_);
  float* hs_b = allocF((size_t)LB_ * B_ * D_); float* ctx_b = allocF((size_t)LB_ * B_ * E_);
  float* hs_g = allocF((size_t)LB_ * B_ * D_); float* ctx_g = allocF((size_t)LB_ * B_ * E_);
  float* cf   = allocF((size_t)LC_ * B_ * D_);

  half_t* catA_c  = allocH((size_t)LC_ * B_ * 2176);
  float*  logit_c = allocF((size_t)LC_ * B_ * 1024);
  half_t* lc16    = allocH((size_t)LC_ * B_ * 512);
  float*  t32 = allocF((size_t)LC_ * B_ * 1024);
  half_t* t16 = allocH((size_t)LC_ * B_ * 1024);
  float*  u32 = allocF((size_t)LC_ * B_ * 1024);

  half_t* hsb16  = allocH((size_t)LB_ * B_ * 1024);
  float*  g32    = allocF((size_t)LB_ * B_ * 1024);
  half_t* catA_b = allocH((size_t)LB_ * B_ * 5120);
  float*  logit_b = allocF((size_t)LB_ * B_ * 2048);
  half_t* lb16    = allocH((size_t)LB_ * B_ * 1024);

  // ---------------- pack weights (f32 -> f16 fragment layout) ----------------
  auto clr = [&](size_t halves) { hipMemsetAsync(staging, 0, halves * sizeof(half_t), stream); };

  clr((size_t)1024 * 1024); padc(stream, F(42), 1024, 1024, staging, 1024, 0, 0);
  packW(stream, staging, cinit_p, 1024, 1024);
  clr((size_t)1024 * 1024); padc(stream, F(44), 1024, 1024, staging, 1024, 0, 0);
  packW(stream, staging, binit_p, 1024, 1024);

  for (int k = 0; k < 3; ++k) {
    int b0 = sbase[k];
    // Wx: [einPad, 3072] = 1_W (cols 0..2047) | 1_Wx (cols 2048..3071)
    clr((size_t)einP[k] * 3072);
    padc(stream, F(b0 + 2), ein[k], 2048, staging, 3072, 0, 0);
    padc(stream, F(b0 + 0), ein[k], 1024, staging, 3072, 0, 2048);
    packW(stream, staging, Wx_p[k], einP[k], 3072);
    copyf(stream, xbias[k], F(b0 + 3), 2048);
    copyf(stream, xbias[k] + 2048, F(b0 + 1), 1024);
    // W1: [1024, 3072] = 1_U | 1_Ux
    clr((size_t)1024 * 3072);
    padc(stream, F(b0 + 4), 1024, 2048, staging, 3072, 0, 0);
    padc(stream, F(b0 + 5), 1024, 1024, staging, 3072, 0, 2048);
    packW(stream, staging, W1_p[k], 1024, 3072);
    // W2: [2048, 4096] = [2_U;2_Wc] | [2_Ux;0] | [0;2_Wcx]
    clr((size_t)2048 * 4096);
    padc(stream, F(b0 + 6),  1024, 2048, staging, 4096, 0,    0);
    padc(stream, F(b0 + 8),  1024, 2048, staging, 4096, 1024, 0);
    padc(stream, F(b0 + 9),  1024, 1024, staging, 4096, 0,    2048);
    padc(stream, F(b0 + 11), 1024, 1024, staging, 4096, 1024, 3072);
    packW(stream, staging, W2_p[k], 2048, 4096);
    hipMemsetAsync(bias2[k], 0, 4096 * sizeof(float), stream);
    copyf(stream, bias2[k], F(b0 + 7), 2048);
    copyf(stream, bias2[k] + 2048, F(b0 + 10), 1024);
  }

  clr((size_t)2176 * 1024);
  padc(stream, F(46), 1024, 1024, staging, 1024, 0,    0);
  padc(stream, F(48),  128, 1024, staging, 1024, 1024, 0);
  padc(stream, F(49), 1024, 1024, staging, 1024, 1152, 0);
  packW(stream, staging, cc_p, 2176, 1024);

  clr((size_t)512 * 4000); padc(stream, F(50), 512, 4000, staging, 4000, 0, 0);
  packW(stream, staging, ccout_p, 512, 4000);
  clr((size_t)1024 * 1024); padc(stream, F(52), 1024, 1024, staging, 1024, 0, 0);
  packW(stream, staging, b2c1_p, 1024, 1024);
  clr((size_t)1024 * 1024); padc(stream, F(54), 1024, 1024, staging, 1024, 0, 0);
  packW(stream, staging, b2c2_p, 1024, 1024);
  clr((size_t)1024 * 1024); padc(stream, F(56), 1024, 1024, staging, 1024, 0, 0);
  packW(stream, staging, comb_p, 1024, 1024);

  clr((size_t)5120 * 2048);
  padc(stream, F(58), 2048, 2048, staging, 2048, 0,    0);
  padc(stream, F(60), 1000, 2048, staging, 2048, 2048, 0);
  padc(stream, F(61), 2048, 2048, staging, 2048, 3072, 0);
  packW(stream, staging, cb_p, 5120, 2048);

  clr((size_t)1024 * 2000); padc(stream, F(62), 1024, 2000, staging, 2000, 0, 0);
  packW(stream, staging, cbout_p, 1024, 2000);

  // ---------------- embeddings, mean, init states ----------------
  k_embed_c<<<(LC_ * B_ * EC_ + 255) / 256, 256, 0, stream>>>(F(4), ysc, embc16);
  k_embed_b<<<(LB_ * B_ * 1024 + 255) / 256, 256, 0, stream>>>(F(5), ysb, embb16);
  k_encmean<<<64, 256, 0, stream>>>(hs_pad, hlens, mean32, mean16);
  run_gemm_m16(stream, mean16, cinit_p, F(43), s0c, 1024, 1024, 1024);
  run_gemm_m16(stream, mean16, binit_p, F(45), s0b, 1024, 1024, 1024);

  // xW precompute (time-parallel WMMA GEMMs)
  run_gemm(stream, embc16, 128,  Wx_p[0], xbias[0], xWc, 3072, LC_ * B_, 3072, 128);
  run_gemm(stream, embb16, 1024, Wx_p[1], xbias[1], xWb, 3072, LB_ * B_, 3072, 1024);
  run_gemm(stream, embb16, 1024, Wx_p[2], xbias[2], xWg, 3072, LB_ * B_, 3072, 1024);

  // ---------------- char decoder ----------------
  run_stack(stream, LC_, hs_pad, (long)T_ * E_, E_, T_, hlens,
            xWc, W1_p[0], W2_p[0], bias2[0], s0c, state, sA16, hc16,
            Y1, Y2, h1buf, scores, alphab, hs_c, ctx_c);

  // char classifier A = [hs_c | emb_c | ctx_c]
  {
    long n = (long)LC_ * B_ * 2176;
    k_cat_c<<<(int)((n + 255) / 256), 256, 0, stream>>>(hs_c, embc16, ctx_c, catA_c);
  }
  // b2c residual branch -> cf (memory for char2bpe attention)
  run_gemm(stream, catA_c, 2176, b2c1_p, F(53), t32, 1024, LC_ * B_, 1024, 1024);
  k_relu16<<<(int)(((long)LC_ * B_ * 1024 + 255) / 256), 256, 0, stream>>>(t32, t16, (long)LC_ * B_ * 1024);
  run_gemm(stream, t16, 1024, b2c2_p, F(55), u32, 1024, LC_ * B_, 1024, 1024);
  k_add2<<<(int)(((long)LC_ * B_ * 1024 + 255) / 256), 256, 0, stream>>>(cf, hs_c, u32, (long)LC_ * B_ * 1024);

  // char logits -> d_out[0 .. B*LC*VC)
  float* outC = (float*)d_out;
  run_gemm(stream, catA_c, 2176, cc_p, F(47), logit_c, 1024, LC_ * B_, 1024, 2176);
  k_maxout2<<<(int)(((long)LC_ * B_ * 512 + 255) / 256), 256, 0, stream>>>(logit_c, lc16, LC_ * B_, 512);
  run_gemm(stream, lc16, 512, ccout_p, F(51), outC, VC_, LC_ * B_, VC_, 512, 1, B_, LC_);

  // ---------------- bpe + char2bpe decoders ----------------
  run_stack(stream, LB_, hs_pad, (long)T_ * E_, E_, T_, hlens,
            xWb, W1_p[1], W2_p[1], bias2[1], s0b, state, sA16, hc16,
            Y1, Y2, h1buf, scores, alphab, hs_b, ctx_b);
  // char2bpe attends cf: mem[b,t,d] = cf[(t*B+b)*D + d]
  run_stack(stream, LB_, cf, (long)D_, (long)B_ * D_, LC_, nullptr,
            xWg, W1_p[2], W2_p[2], bias2[2], s0b, state, sA16, hc16,
            Y1, Y2, h1buf, scores, alphab, hs_g, ctx_g);

  // deep fusion + bpe classifier -> d_out[B*LC*VC ..)
  k_cast16<<<(int)(((long)LB_ * B_ * 1024 + 255) / 256), 256, 0, stream>>>(hs_b, hsb16, (long)LB_ * B_ * 1024);
  run_gemm(stream, hsb16, 1024, comb_p, F(57), g32, 1024, LB_ * B_, 1024, 1024);
  {
    long n = (long)LB_ * B_ * 5120;
    k_cat_b<<<(int)((n + 255) / 256), 256, 0, stream>>>(hs_b, g32, hs_g, embb16, ctx_b, ctx_g, catA_b);
  }
  run_gemm(stream, catA_b, 5120, cb_p, F(59), logit_b, 2048, LB_ * B_, 2048, 5120);
  k_maxout2<<<(int)(((long)LB_ * B_ * 1024 + 255) / 256), 256, 0, stream>>>(logit_b, lb16, LB_ * B_, 1024);
  float* outB = outC + (size_t)B_ * LC_ * VC_;
  run_gemm(stream, lb16, 1024, cbout_p, F(63), outB, VB_, LB_ * B_, VB_, 1024, 1, B_, LB_);
}
